// PDG2Seq_Encoder_22969485099153
// MI455X (gfx1250) — compile-verified, matrix operand order checked by reading, not run on hardware
//
#include <hip/hip_runtime.h>

#define Bq    16
#define Tq    12
#define Nq    1024
#define EMBq  10
#define DINq  21
#define DOUTq 64
#define FINq  85    // DIN + DOUT
#define FPq   96    // padded feature cols for the A@X GEMM
#define KFq   170   // CHEB_K * FIN
#define KFPq  192   // K padded to multiple of 32 for WMMA

typedef __attribute__((ext_vector_type(16))) __bf16 v16bf;
typedef __attribute__((ext_vector_type(8)))  float  v8f;
typedef __attribute__((ext_vector_type(4)))  unsigned v4u;
typedef __attribute__((ext_vector_type(8)))  int    v8i;
typedef __attribute__((ext_vector_type(4)))  int    v4i;

union FragU { unsigned u[8]; v16bf v; };

__device__ __forceinline__ unsigned short f2bf(float f) {
  unsigned u = __builtin_bit_cast(unsigned, f);
  u += 0x7FFFu + ((u >> 16) & 1u);   // round-to-nearest-even
  return (unsigned short)(u >> 16);
}

__device__ __forceinline__ v8f wmma_bf(const FragU& A, const FragU& B, v8f C) {
  return __builtin_amdgcn_wmma_f32_16x16x32_bf16(
      /*neg_a=*/false, A.v, /*neg_b=*/false, B.v,
      /*c_mod=*/(short)0, C, /*reuse_a=*/false, /*reuse_b=*/false);
}

// ---- Tensor Data Mover: 2-D tile (2-byte elements) global -> LDS ----
__device__ __forceinline__ unsigned lds_off(const void* p) {
  return (unsigned)(size_t)p;  // low 32 bits of generic addr = LDS byte offset
}

__device__ __forceinline__ void wait_tensorcnt0() {
#if __has_builtin(__builtin_amdgcn_s_wait_tensorcnt)
  __builtin_amdgcn_s_wait_tensorcnt(0);
#else
  asm volatile("s_wait_tensorcnt 0x0" ::: "memory");
#endif
}

// D# per CDNA5 ISA 8.3/8.4: group0 = {count/type, lds_addr, global_addr}, group1 = dims/strides.
__device__ __forceinline__ void tdm_load_2d(const void* gsrc, unsigned ldsaddr,
                                            unsigned tensor_d0, unsigned tensor_d1,
                                            unsigned tile_d0, unsigned tile_d1,
                                            unsigned long long d0_stride) {
  unsigned long long ga = (unsigned long long)gsrc;
  v4u g0;
  g0[0] = 1u;                                   // count=1, user descriptor, no gather
  g0[1] = ldsaddr;                              // lds_addr (bytes)
  g0[2] = (unsigned)ga;                         // global_addr[31:0]
  g0[3] = (unsigned)((ga >> 32) & 0x01FFFFFFu) | (2u << 30);  // addr[56:32], type=2
  v8i g1;
  g1[0] = (int)(1u << 16);                      // workgroup_mask=0, data_size=1 (2 bytes)
  g1[1] = (int)(tensor_d0 << 16);               // tensor_dim0[15:0] at bits 63:48
  g1[2] = (int)((tensor_d0 >> 16) | (tensor_d1 << 16));   // dim0 hi | dim1 lo
  g1[3] = (int)((tensor_d1 >> 16) | (tile_d0 << 16));     // dim1 hi | tile_dim0
  g1[4] = (int)(tile_d1 & 0xFFFFu);             // tile_dim1 (tile_dim2 = 0 -> 2-D)
  g1[5] = (int)(unsigned)d0_stride;             // tensor_dim0_stride[31:0]
  g1[6] = (int)(unsigned)((d0_stride >> 32) & 0xFFFFu);   // stride hi (dim1_stride = 0)
  g1[7] = 0;
  v4i gz = {0, 0, 0, 0};
#if defined(__clang_major__) && (__clang_major__ >= 23)
  v8i gz8 = {0, 0, 0, 0, 0, 0, 0, 0};
  __builtin_amdgcn_tensor_load_to_lds(g0, g1, gz, gz, gz8, 0);
#else
  __builtin_amdgcn_tensor_load_to_lds(g0, g1, gz, gz, 0);
#endif
}

// A fragment: 16x32 bf16 tile from row-major [M x K] (ld = ldk, even; base 4B-aligned).
// Each VGPR = one aligned b32 load of the (k,k+1) pair.
template <typename P>
__device__ __forceinline__ void frag_a(FragU& F, P p, int ldk, int row0, int k0, int lane) {
  int m = lane & 15, hi = lane >> 4;
#pragma unroll
  for (int v = 0; v < 8; ++v) {
    int k = (v < 4) ? (hi * 8 + v * 2) : (16 + hi * 8 + (v - 4) * 2);
    F.u[v] = *(const unsigned*)&p[(size_t)(row0 + m) * ldk + k0 + k];
  }
}

// B fragment from PAIR-INTERLEAVED [K/2][N][2] storage: one aligned b32 load per VGPR.
template <typename P>
__device__ __forceinline__ void frag_b_pairs(FragU& F, P p, int ldn, int k0, int col0, int lane) {
  int n = lane & 15, hi = lane >> 4;
#pragma unroll
  for (int v = 0; v < 8; ++v) {
    int k = (v < 4) ? (hi * 8 + v * 2) : (16 + hi * 8 + (v - 4) * 2);
    F.u[v] = *(const unsigned*)&p[((size_t)((k0 + k) >> 1) * ldn + col0 + n) * 2];
  }
}

// ---------------- setup kernels ----------------

__global__ void k_embed(const float* __restrict__ x, const float* __restrict__ Wtod,
                        const float* __restrict__ btod, const float* __restrict__ Wdow,
                        const float* __restrict__ bdow, float* __restrict__ cur) {
  int i = blockIdx.x * blockDim.x + threadIdx.x;
  if (i >= Bq * Tq * Nq) return;
  float f0 = x[i * 3 + 0];
  int tod = (int)x[i * 3 + 1];
  int dow = (int)x[i * 3 + 2];
  float* o = cur + (size_t)i * DINq;
  o[0] = f0;
#pragma unroll
  for (int d = 0; d < EMBq; ++d) {
    o[1 + d]        = Wtod[tod * EMBq + d] + btod[d];
    o[1 + EMBq + d] = Wdow[dow * EMBq + d] + bdow[d];
  }
}

// Wout panel per node n is [KFP/2][O][2] pair-interleaved (contiguous KFP*O elements).
__global__ void k_pool_w(const float* __restrict__ E, const float* __restrict__ Wpool,
                         unsigned short* __restrict__ Wout, int O) {
  int n = blockIdx.x, kf = blockIdx.y, o = threadIdx.x;
  float acc = 0.f;
  if (kf < KFq) {
    int k = (kf >= FINq) ? 1 : 0;
    int f = kf - k * FINq;
    const float* wp = Wpool + (size_t)(k * FINq + f) * O + o;
#pragma unroll
    for (int d = 0; d < EMBq; ++d)
      acc += E[n * EMBq + d] * wp[(size_t)d * 2 * FINq * O];
  }
  Wout[(size_t)n * KFPq * O + (size_t)((kf >> 1) * O + o) * 2 + (kf & 1)] = f2bf(acc);
}

__global__ void k_pool_b(const float* __restrict__ E, const float* __restrict__ bpool,
                         float* __restrict__ bout, int O) {
  int n = blockIdx.x, o = threadIdx.x;
  float acc = 0.f;
#pragma unroll
  for (int d = 0; d < EMBq; ++d) acc += E[n * EMBq + d] * bpool[d * O + o];
  bout[(size_t)n * O + o] = acc;
}

// float [K x Ncols] -> bf16 pair-interleaved [K/2][Ncols][2]
__global__ void k_cvt_pairs(const float* __restrict__ src, unsigned short* __restrict__ dst,
                            int K, int Ncols) {
  int i = blockIdx.x * blockDim.x + threadIdx.x;
  if (i >= K * Ncols) return;
  int r = i / Ncols, c = i % Ncols;
  dst[((r >> 1) * Ncols + c) * 2 + (r & 1)] = f2bf(src[i]);
}

__global__ void k_copy(const float* __restrict__ src, float* __restrict__ dst, int count) {
  int i = blockIdx.x * blockDim.x + threadIdx.x;
  if (i < count) dst[i] = src[i];
}

// ---------------- per-timestep kernels ----------------

// xinB[b] is [Nq/2][FPq][2] pair-interleaved over the node (K) dimension.
__global__ void k_build_xin(const float* __restrict__ cur, const float* __restrict__ h,
                            const float* __restrict__ zrF, int t, int mode,
                            unsigned short* __restrict__ xinB) {
  int i = blockIdx.x * blockDim.x + threadIdx.x;
  if (i >= Bq * Nq * FPq) return;
  int f = i % FPq; int bn = i / FPq; int m = bn % Nq; int b = bn / Nq;
  float val = 0.f;
  if (f < DINq) {
    val = cur[(((size_t)b * Tq + t) * Nq + m) * DINq + f];
  } else if (f < FINq) {
    float hv = h[((size_t)b * Nq + m) * DOUTq + (f - DINq)];
    if (mode) hv *= zrF[((size_t)b * Nq + m) * 2 * DOUTq + (f - DINq)];  // z = zr[:, :64]
    val = hv;
  }
  xinB[(size_t)b * Nq * FPq + (size_t)((m >> 1) * FPq + f) * 2 + (m & 1)] = f2bf(val);
}

// A[b] = softmax(relu(d1 d2^T)) rows, stored bf16 row-major. One wave per row; E cached in LDS.
__global__ void __launch_bounds__(256) k_adj(const float* __restrict__ E,
                                             const float* __restrict__ ne_tod,
                                             const float* __restrict__ ne_dow, int t,
                                             unsigned short* __restrict__ Abf) {
  __shared__ float Es[Nq * EMBq];  // 40 KB
  int b = blockIdx.x, rb = blockIdx.y;
  for (int i = threadIdx.x; i < Nq * EMBq; i += 256) Es[i] = E[i];
  __syncthreads();
  int wave = threadIdx.x >> 5, lane = threadIdx.x & 31;
  int r = rb * 8 + wave;
  float rv[EMBq];
#pragma unroll
  for (int d = 0; d < EMBq; ++d) {
    float g = ne_tod[(b * Tq + t) * EMBq + d] * ne_dow[(b * Tq + t) * EMBq + d];
    rv[d] = Es[r * EMBq + d] * g;
  }
  float vals[32];
  float vmax = 0.f;
#pragma unroll 4
  for (int j = 0; j < 32; ++j) {
    int m = j * 32 + lane;
    float dot = 0.f;
#pragma unroll
    for (int d = 0; d < EMBq; ++d) dot += Es[m * EMBq + d] * rv[d];
    float vr = dot > 0.f ? dot : 0.f;   // relu
    vals[j] = vr;
    vmax = fmaxf(vmax, vr);
  }
  for (int off = 16; off; off >>= 1) vmax = fmaxf(vmax, __shfl_xor(vmax, off, 32));
  float s = 0.f;
#pragma unroll 4
  for (int j = 0; j < 32; ++j) { vals[j] = __expf(vals[j] - vmax); s += vals[j]; }
  for (int off = 16; off; off >>= 1) s += __shfl_xor(s, off, 32);
  float inv = 1.f / s;
#pragma unroll 4
  for (int j = 0; j < 32; ++j) {
    int m = j * 32 + lane;
    Abf[((size_t)b * Nq + r) * Nq + m] = f2bf(vals[j] * inv);
  }
}

// Ax[b] (1024x96 f32) = A[b] (1024x1024 bf16) @ xinB[b] (pair-interleaved 1024x96 bf16).
// 128 rows x 96 cols per block; 6 KB X chunks TDM double-buffered into LDS.
__global__ void __launch_bounds__(256) k_gemm_ax(const unsigned short* __restrict__ Abf,
                                                 const unsigned short* __restrict__ xinB,
                                                 float* __restrict__ Ax) {
  __shared__ alignas(16) unsigned short Xs[2][32 * FPq];  // double-buffered X chunk
  int b = blockIdx.x;
  int wave = threadIdx.x >> 5, lane = threadIdx.x & 31;
  int r0 = blockIdx.y * 128 + wave * 16;
  const unsigned short* Ab = Abf + (size_t)b * Nq * Nq;
  const unsigned short* Xb = xinB + (size_t)b * Nq * FPq;
  bool issuer = (wave == 0);
  v8f zero = {};
  v8f acc[6];
#pragma unroll
  for (int ct = 0; ct < 6; ++ct) acc[ct] = zero;

  if (issuer)  // prologue: chunk 0 -> buffer 0  (chunk = 16 pair-rows x (2*FPq) elems, contiguous)
    tdm_load_2d(Xb, lds_off(&Xs[0][0]), 2 * FPq, Nq / 2, 2 * FPq, 16, 2 * FPq);

  for (int i = 0; i < Nq / 32; ++i) {
    if (issuer) wait_tensorcnt0();
    __syncthreads();                      // chunk i visible to all waves
    if (issuer && (i + 1 < Nq / 32))      // prefetch chunk i+1 into the other buffer
      tdm_load_2d(Xb + (size_t)(i + 1) * 32 * FPq, lds_off(&Xs[(i + 1) & 1][0]),
                  2 * FPq, Nq / 2, 2 * FPq, 16, 2 * FPq);
    FragU A;
    frag_a(A, Ab, Nq, r0, i * 32, lane);  // aligned b32 loads from L2
    const unsigned short* Xc = &Xs[i & 1][0];
#pragma unroll
    for (int ct = 0; ct < 6; ++ct) {
      FragU Bf;
      frag_b_pairs(Bf, Xc, FPq, 0, ct * 16, lane);  // single ds_load_b32 per VGPR
      acc[ct] = wmma_bf(A, Bf, acc[ct]);
    }
    __syncthreads();                      // reads done before buffer reuse
  }
  int mrow = (lane < 16) ? 0 : 8, nn = lane & 15;
#pragma unroll
  for (int ct = 0; ct < 6; ++ct)
#pragma unroll
    for (int v = 0; v < 8; ++v)
      Ax[((size_t)b * Nq + r0 + v + mrow) * FPq + ct * 16 + nn] = acc[ct][v];
}

// Per-node einsum via WMMA: C(16=B x O) = X(16 x 192) @ W_n(192 x O); W_n TDM-staged in LDS.
// MODE 0: sigmoid -> zrF (O=128).  MODE 1: tanh, h_new = r*h + (1-r)*hc -> h and seq (O=64).
template <int O, int MODE>
__global__ void k_gconv(const unsigned short* __restrict__ xinB, const float* __restrict__ Ax,
                        const unsigned short* __restrict__ Wn, const float* __restrict__ bias,
                        const float* __restrict__ zrF_in, const float* __restrict__ h_in,
                        float* __restrict__ out0, float* __restrict__ seq, int t) {
  __shared__ alignas(16) unsigned short Xs[16 * KFPq];   // 6 KB, rows = batch (A-matrix)
  __shared__ alignas(16) unsigned short Ws[KFPq * O];    // pair-interleaved weight panel
  int n = blockIdx.x;
  int wave = threadIdx.x >> 5, lane = threadIdx.x & 31;
  const unsigned short* Wp = Wn + (size_t)n * KFPq * O;
  if (wave == 0)  // DMA the whole per-node weight panel while X is built
    tdm_load_2d(Wp, lds_off(&Ws[0]), 2 * O, KFPq / 2, 2 * O, KFPq / 2, 2 * O);
  for (int i = threadIdx.x; i < 16 * KFPq; i += blockDim.x) {
    int b = i / KFPq, kf = i % KFPq;
    unsigned short v;
    if (kf < FINq)
      v = xinB[(size_t)b * Nq * FPq + (size_t)((n >> 1) * FPq + kf) * 2 + (n & 1)];  // k=0 (x)
    else if (kf < KFq)
      v = f2bf(Ax[((size_t)b * Nq + n) * FPq + (kf - FINq)]);                        // k=1 (A@x)
    else
      v = 0;
    Xs[i] = v;
  }
  if (wave == 0) wait_tensorcnt0();
  __syncthreads();
  int c0 = wave * 16;
  v8f acc = {};
  for (int k0 = 0; k0 < KFPq; k0 += 32) {
    FragU A, B;
    frag_a(A, (const unsigned short*)Xs, KFPq, 0, k0, lane);
    frag_b_pairs(B, (const unsigned short*)Ws, O, k0, c0, lane);
    acc = wmma_bf(A, B, acc);
  }
  int brow = (lane < 16) ? 0 : 8;
  int o = c0 + (lane & 15);
#pragma unroll
  for (int v = 0; v < 8; ++v) {
    int b = v + brow;
    float val = acc[v] + bias[(size_t)n * O + o];
    if (MODE == 0) {
      out0[((size_t)b * Nq + n) * 2 * DOUTq + o] = 1.f / (1.f + __expf(-val));
    } else {
      float hc = tanhf(val);
      float r = zrF_in[((size_t)b * Nq + n) * 2 * DOUTq + DOUTq + o];
      float hold = h_in[((size_t)b * Nq + n) * DOUTq + o];
      float hnew = r * hold + (1.f - r) * hc;
      out0[((size_t)b * Nq + n) * DOUTq + o] = hnew;
      seq[(((size_t)b * Tq + t) * Nq + n) * DOUTq + o] = hnew;
    }
  }
}

// ---------------- temporal attention (per b,n; T=12 padded to 16), in-place on d_out ----------------
__global__ void __launch_bounds__(128) k_attn(float* __restrict__ seqout,
                                              const unsigned short* __restrict__ Wq,
                                              const unsigned short* __restrict__ Wk,
                                              const unsigned short* __restrict__ Wv,
                                              const float* __restrict__ bQ,
                                              const float* __restrict__ bK,
                                              const float* __restrict__ bV) {
  int bn = blockIdx.x; int b = bn / Nq; int n = bn % Nq;
  __shared__ alignas(16) unsigned short Xs[16 * 64], Qs[16 * 64], Ks[16 * 64];
  __shared__ alignas(16) unsigned short VsP[16 * 64 * 2];  // pair-interleaved V, K rows 0..31
  __shared__ float Ss[16 * 16];
  __shared__ alignas(16) unsigned short Sb[16 * 32];
  for (int i = threadIdx.x; i < 16 * 64; i += 128) {
    int tt = i >> 6, d = i & 63;
    float v = (tt < Tq) ? seqout[(((size_t)b * Tq + tt) * Nq + n) * DOUTq + d] : 0.f;
    Xs[i] = f2bf(v);
    VsP[1024 + i] = 0;  // zero pad pairs 8..15 (K rows 16..31)
  }
  __syncthreads();
  int wave = threadIdx.x >> 5, lane = threadIdx.x & 31;
  int c0 = wave * 16;
  {  // Q,K,V projections for this wave's 16-col tile
    v8f aq = {}, ak = {}, av = {};
    for (int k0 = 0; k0 < 64; k0 += 32) {
      FragU A, Fq, Fk, Fv;
      frag_a(A, (const unsigned short*)Xs, 64, 0, k0, lane);
      frag_b_pairs(Fq, Wq, 64, k0, c0, lane);
      frag_b_pairs(Fk, Wk, 64, k0, c0, lane);
      frag_b_pairs(Fv, Wv, 64, k0, c0, lane);
      aq = wmma_bf(A, Fq, aq);
      ak = wmma_bf(A, Fk, ak);
      av = wmma_bf(A, Fv, av);
    }
    int trow = (lane < 16) ? 0 : 8, d = c0 + (lane & 15);
#pragma unroll
    for (int v = 0; v < 8; ++v) {
      int tt = v + trow;
      Qs[tt * 64 + d] = f2bf(aq[v] + bQ[d]);
      Ks[tt * 64 + d] = f2bf(ak[v] + bK[d]);
      VsP[((tt >> 1) * 64 + d) * 2 + (tt & 1)] = f2bf(av[v] + bV[d]);
    }
  }
  __syncthreads();
  if (wave == 0) {  // S = Q K^T / 8  (K^T pairs are contiguous along d -> b32 loads)
    v8f s = {};
    for (int k0 = 0; k0 < 64; k0 += 32) {
      FragU A, B;
      frag_a(A, (const unsigned short*)Qs, 64, 0, k0, lane);
      int sc = lane & 15, hi = lane >> 4;
#pragma unroll
      for (int v = 0; v < 8; ++v) {
        int k = (v < 4) ? (hi * 8 + v * 2) : (16 + hi * 8 + (v - 4) * 2);
        B.u[v] = *(const unsigned*)&Ks[sc * 64 + k0 + k];
      }
      s = wmma_bf(A, B, s);
    }
    int trow = (lane < 16) ? 0 : 8, sc = lane & 15;
#pragma unroll
    for (int v = 0; v < 8; ++v) Ss[(v + trow) * 16 + sc] = s[v] * 0.125f;
  }
  __syncthreads();
  if (threadIdx.x < 16) {  // row softmax over first 12 cols
    int tt = threadIdx.x;
    if (tt < Tq) {
      float mx = -1e30f;
      for (int s2 = 0; s2 < Tq; ++s2) mx = fmaxf(mx, Ss[tt * 16 + s2]);
      float ex[Tq]; float sum = 0.f;
      for (int s2 = 0; s2 < Tq; ++s2) { ex[s2] = __expf(Ss[tt * 16 + s2] - mx); sum += ex[s2]; }
      float inv = 1.f / sum;
      for (int s2 = 0; s2 < 32; ++s2)
        Sb[tt * 32 + s2] = (s2 < Tq) ? f2bf(ex[s2] * inv) : (unsigned short)0;
    } else {
      for (int s2 = 0; s2 < 32; ++s2) Sb[tt * 32 + s2] = 0;
    }
  }
  __syncthreads();
  {  // Z = S @ V (K=16 zero-padded to 32; V pair-interleaved, pad rows pre-zeroed)
    v8f z = {};
    FragU A, B;
    frag_a(A, (const unsigned short*)Sb, 32, 0, 0, lane);
    frag_b_pairs(B, (const unsigned short*)VsP, 64, 0, c0, lane);
    z = wmma_bf(A, B, z);
    int trow = (lane < 16) ? 0 : 8, nn = lane & 15;
#pragma unroll
    for (int v = 0; v < 8; ++v) {
      int tt = v + trow;
      if (tt < Tq) seqout[(((size_t)b * Tq + tt) * Nq + n) * DOUTq + c0 + nn] = z[v];
    }
  }
}

extern "C" void kernel_launch(void* const* d_in, const int* in_sizes, int n_in,
                              void* d_out, int out_size, void* d_ws, size_t ws_size,
                              hipStream_t stream) {
  (void)in_sizes; (void)n_in; (void)out_size; (void)ws_size;
  const float* x        = (const float*)d_in[0];
  const float* init_st  = (const float*)d_in[1];
  const float* ne_tod   = (const float*)d_in[2];
  const float* ne_dow   = (const float*)d_in[3];
  const float* node_emb = (const float*)d_in[4];
  const float* Wtod = (const float*)d_in[5];  const float* btod = (const float*)d_in[6];
  const float* Wdow = (const float*)d_in[7];  const float* bdow = (const float*)d_in[8];
  const float* WQ = (const float*)d_in[9];    const float* bQ = (const float*)d_in[10];
  const float* WK = (const float*)d_in[11];   const float* bK = (const float*)d_in[12];
  const float* WV = (const float*)d_in[13];   const float* bV = (const float*)d_in[14];
  const float* Wpool_g = (const float*)d_in[15]; const float* bpool_g = (const float*)d_in[16];
  const float* Wpool_c = (const float*)d_in[17]; const float* bpool_c = (const float*)d_in[18];

  float* out = (float*)d_out;                                   // (B,T,N,64)
  float* hlast = out + (size_t)Bq * Tq * Nq * DOUTq;            // (B,N,64)

  // workspace carve-up
  uintptr_t w = (uintptr_t)d_ws;
  auto take = [&](size_t bytes) -> void* {
    void* p = (void*)w; w += (bytes + 255) & ~(size_t)255; return p;
  };
  float* cur           = (float*)take((size_t)Bq * Tq * Nq * DINq * 4);
  unsigned short* WgB  = (unsigned short*)take((size_t)Nq * KFPq * 128 * 2);
  float* bgF           = (float*)take((size_t)Nq * 128 * 4);
  unsigned short* WcB  = (unsigned short*)take((size_t)Nq * KFPq * 64 * 2);
  float* bcF           = (float*)take((size_t)Nq * 64 * 4);
  float* h             = (float*)take((size_t)Bq * Nq * DOUTq * 4);
  unsigned short* Abf  = (unsigned short*)take((size_t)Bq * Nq * Nq * 2);
  unsigned short* xinB = (unsigned short*)take((size_t)Bq * Nq * FPq * 2);
  float* Ax            = (float*)take((size_t)Bq * Nq * FPq * 4);
  float* zrF           = (float*)take((size_t)Bq * Nq * 128 * 4);
  unsigned short* WQb  = (unsigned short*)take(64 * 64 * 2);
  unsigned short* WKb  = (unsigned short*)take(64 * 64 * 2);
  unsigned short* WVb  = (unsigned short*)take(64 * 64 * 2);

  // ---- setup ----
  k_embed<<<(Bq * Tq * Nq + 255) / 256, 256, 0, stream>>>(x, Wtod, btod, Wdow, bdow, cur);
  k_pool_w<<<dim3(Nq, KFPq), 128, 0, stream>>>(node_emb, Wpool_g, WgB, 128);
  k_pool_w<<<dim3(Nq, KFPq), 64, 0, stream>>>(node_emb, Wpool_c, WcB, 64);
  k_pool_b<<<Nq, 128, 0, stream>>>(node_emb, bpool_g, bgF, 128);
  k_pool_b<<<Nq, 64, 0, stream>>>(node_emb, bpool_c, bcF, 64);
  k_cvt_pairs<<<(64 * 64 + 255) / 256, 256, 0, stream>>>(WQ, WQb, 64, 64);
  k_cvt_pairs<<<(64 * 64 + 255) / 256, 256, 0, stream>>>(WK, WKb, 64, 64);
  k_cvt_pairs<<<(64 * 64 + 255) / 256, 256, 0, stream>>>(WV, WVb, 64, 64);
  hipMemcpyAsync(h, init_st, (size_t)Bq * Nq * DOUTq * 4, hipMemcpyDeviceToDevice, stream);

  // ---- GRU scan over T ----
  for (int t = 0; t < Tq; ++t) {
    k_adj<<<dim3(Bq, Nq / 8), 256, 0, stream>>>(node_emb, ne_tod, ne_dow, t, Abf);
    // gate path: xin = [cur_t, h]
    k_build_xin<<<(Bq * Nq * FPq + 255) / 256, 256, 0, stream>>>(cur, h, zrF, t, 0, xinB);
    k_gemm_ax<<<dim3(Bq, Nq / 128), 256, 0, stream>>>(Abf, xinB, Ax);
    k_gconv<128, 0><<<Nq, 256, 0, stream>>>(xinB, Ax, WgB, bgF, nullptr, nullptr, zrF, nullptr, t);
    // candidate path: xin = [cur_t, z*h]
    k_build_xin<<<(Bq * Nq * FPq + 255) / 256, 256, 0, stream>>>(cur, h, zrF, t, 1, xinB);
    k_gemm_ax<<<dim3(Bq, Nq / 128), 256, 0, stream>>>(Abf, xinB, Ax);
    k_gconv<64, 1><<<Nq, 128, 0, stream>>>(xinB, Ax, WcB, bcF, zrF, h, h, out, t);
  }

  // ---- temporal attention (in-place on d_out seq region) ----
  k_attn<<<Bq * Nq, 128, 0, stream>>>(out, WQb, WKb, WVb, bQ, bK, bV);

  // ---- h_last output ----
  k_copy<<<(Bq * Nq * DOUTq + 255) / 256, 256, 0, stream>>>(h, hlast, Bq * Nq * DOUTq);
}
